// CasualSelfAttention_62019327754273
// MI455X (gfx1250) — matrix operand order repaired and checked by reading, not compiled, not run on hardware
//
#include <hip/hip_runtime.h>
#include <hip/hip_bf16.h>

// ---------------------------------------------------------------------------
// Causal self-attention (column softmax) for B=4, T=2048, C=768, H=12, hd=64
// bf16 WMMA (v_wmma_f32_16x16x32_bf16) everywhere, f32 accumulate.
// Dense GEMMs: A tiles DMA'd by the Tensor Data Mover, double-buffered LDS,
// one barrier per k-step, TDM/global loads overlapped with WMMA compute.
// Attention: diagonal-peeled column softmax (mask code only on diagonal tiles).
// ---------------------------------------------------------------------------

#define BB 4
#define TT 2048
#define CC 768
#define HH 12
#define HD 64
#define C3 2304          // 3*C
#define MM (BB * TT)     // 8192 rows for the dense GEMMs

typedef __bf16 bhalf;
typedef __attribute__((ext_vector_type(8)))  __bf16 bhalf8;
typedef __attribute__((ext_vector_type(16))) __bf16 bhalf16;
typedef __attribute__((ext_vector_type(8)))  float  floatx8;
typedef __attribute__((ext_vector_type(4)))  float  floatx4;
typedef __attribute__((ext_vector_type(4)))  unsigned int uint32x4;
typedef __attribute__((ext_vector_type(8)))  int    int32x8;
typedef __attribute__((ext_vector_type(4)))  int    int32x4;

static __device__ __forceinline__ floatx8 wmma_bf16(bhalf16 a, bhalf16 b, floatx8 c) {
    // 8 args: (neg_a, A, neg_b, B, c_mod, C, reuse_a, reuse_b)
    return __builtin_amdgcn_wmma_f32_16x16x32_bf16(false, a, false, b, (short)0, c, false, false);
}

static __device__ __forceinline__ bhalf8 ld8(const bhalf* p) {
    return *reinterpret_cast<const bhalf8*>(p);
}

static __device__ __forceinline__ bhalf16 pack16(bhalf8 lo, bhalf8 hi) {
    bhalf16 r;
    reinterpret_cast<bhalf8*>(&r)[0] = lo;
    reinterpret_cast<bhalf8*>(&r)[1] = hi;
    return r;
}

static __device__ __forceinline__ floatx8 zero8() {
    floatx8 z;
#pragma unroll
    for (int i = 0; i < 8; ++i) z[i] = 0.0f;
    return z;
}

// TDM: load a [rows x 32] bf16 tile (row stride K elements in global memory)
// into LDS at lds_byte_addr, rows packed contiguously (32 elems = 64 B apart).
// Descriptor per CDNA5 ISA.
static __device__ __forceinline__ void tdm_load_tile_b16(
    unsigned int lds_byte_addr, const bhalf* gptr, int K, int rows, int cols) {
    const unsigned long long ga = (unsigned long long)(uintptr_t)gptr;

    uint32x4 g0 = {
        1u,                                   // count=1, user descriptor
        lds_byte_addr,                        // lds_addr
        (unsigned int)ga,                     // global_addr[31:0]
        (((unsigned int)(ga >> 32)) & 0x01FFFFFFu) | 0x80000000u  // addr[56:32] | type=2
    };
    int32x8 g1 = {
        (int)(1u << 16),                                        // data_size=1 -> 2 bytes
        (int)(((unsigned int)K & 0xFFFFu) << 16),               // tensor_dim0[15:0]
        (int)((((unsigned int)K >> 16) & 0xFFFFu) |
              (((unsigned int)rows & 0xFFFFu) << 16)),          // dim0 hi | tensor_dim1 lo
        (int)(((unsigned int)cols & 0xFFFFu) << 16),            // tensor_dim1 hi=0 | tile_dim0
        (int)((unsigned int)rows & 0xFFFFu),                    // tile_dim1 | tile_dim2=0
        K,                                                      // tensor_dim0_stride[31:0]
        0,                                                      // stride hi | dim1_stride lo
        0
    };
    int32x4 gz = {0, 0, 0, 0};
#if defined(__clang_major__) && __clang_major__ >= 23
    int32x8 gz8 = {0, 0, 0, 0, 0, 0, 0, 0};
    __builtin_amdgcn_tensor_load_to_lds(g0, g1, gz, gz, gz8, 0);
#else
    __builtin_amdgcn_tensor_load_to_lds(g0, g1, gz, gz, 0);
#endif
}

// ---------------------------------------------------------------------------
// fp32 -> bf16 conversion (vectorized grid-stride, n % 8 == 0)
// ---------------------------------------------------------------------------
__global__ void cvt_f32_bf16_kernel(const float* __restrict__ src,
                                    bhalf* __restrict__ dst, int n) {
    const int stride = gridDim.x * blockDim.x * 8;
    for (int i = (blockIdx.x * blockDim.x + threadIdx.x) * 8; i < n; i += stride) {
        floatx4 a = *reinterpret_cast<const floatx4*>(src + i);
        floatx4 b = *reinterpret_cast<const floatx4*>(src + i + 4);
        bhalf8 o;
#pragma unroll
        for (int j = 0; j < 4; ++j) { o[j] = (bhalf)a[j]; o[4 + j] = (bhalf)b[j]; }
        *reinterpret_cast<bhalf8*>(dst + i) = o;
    }
}

// ---------------------------------------------------------------------------
// Tiled GEMM: out[M,N] = A[M,K] * Bm[K,N] + bias[N]
// Block tile 128x64, 8 waves, each wave owns a 32x32 patch (2x2 WMMA tiles).
// Double-buffered: TDM fills As[next] and global loads prefetch B[next]
// while the current tile's WMMAs run; one barrier per k-step.
// ---------------------------------------------------------------------------
template <int N, int K, bool OUTBF16>
__global__ __launch_bounds__(256) void gemm_bf16_kernel(
    const bhalf* __restrict__ A, const bhalf* __restrict__ Bm,
    const float* __restrict__ bias, void* __restrict__ outv) {

    __shared__ __align__(16) bhalf As[2][128 * 32];  // TDM-filled, row stride 32
    __shared__ __align__(16) bhalf Bs[2][64 * 40];   // transposed [n][k], stride 40

    const int tid   = threadIdx.x;
    const int lane  = tid & 31;
    const int wave  = tid >> 5;
    const int l16   = lane & 15;
    const int halfL = lane >> 4;
    const int wm    = wave & 3;   // 4 M-quadrants of 32 rows
    const int wn    = wave >> 2;  // 2 N-halves of 32 cols

    const int bm = blockIdx.y * 128;
    const int bn = blockIdx.x * 64;

    floatx8 acc[2][2];
#pragma unroll
    for (int i = 0; i < 2; ++i)
#pragma unroll
        for (int j = 0; j < 2; ++j) acc[i][j] = zero8();

    const unsigned int as_lds[2] = {
        (unsigned int)(uintptr_t)(&As[0][0]),
        (unsigned int)(uintptr_t)(&As[1][0])
    };
    const int bk  = tid >> 3;        // 0..31  (k row of B tile)
    const int bnc = (tid & 7) * 8;   // 0..56  (n chunk of B tile)

    // ---- prologue: kick TDM for tile 0, preload B tile 0 into registers ----
    if (wave == 0) tdm_load_tile_b16(as_lds[0], A + (size_t)bm * K, K, 128, 32);
    bhalf8 bv = ld8(Bm + (size_t)bk * N + bn + bnc);

    const int nTiles = K / 32;
    for (int t = 0; t < nTiles; ++t) {
        const int cur = t & 1;

        // stage preloaded B registers into Bs[cur]
#pragma unroll
        for (int j = 0; j < 8; ++j) Bs[cur][(bnc + j) * 40 + bk] = bv[j];

        // As[cur] TDM completion (only wave 0 has an outstanding TENSORcnt),
        // then barrier makes both LDS tiles visible to all waves.
        if (wave == 0) __builtin_amdgcn_s_wait_tensorcnt(0);
        __syncthreads();

        // overlap next tile's data movement with this tile's WMMAs; safe:
        // everyone passed the barrier, so reads of the [next] buffers are done.
        if (t + 1 < nTiles) {
            if (wave == 0)
                tdm_load_tile_b16(as_lds[(t + 1) & 1],
                                  A + (size_t)bm * K + (t + 1) * 32, K, 128, 32);
            const bhalf* gb = Bm + (size_t)((t + 1) * 32 + bk) * N + bn + bnc;
            __builtin_prefetch(gb + 32 * N, 0, 0);  // t+2 hint -> global_prefetch_b8
            bv = ld8(gb);
        }

        // ---- fragments from As[cur] / Bs[cur] ----
        bhalf16 afrag[2], bfrag[2];
#pragma unroll
        for (int i = 0; i < 2; ++i) {
            const int r = wm * 32 + i * 16 + l16;
            bhalf8 c0 = *reinterpret_cast<const bhalf8*>(&As[cur][r * 32 + halfL * 8]);
            bhalf8 c1 = *reinterpret_cast<const bhalf8*>(&As[cur][r * 32 + 16 + halfL * 8]);
            afrag[i] = pack16(c0, c1);
        }
#pragma unroll
        for (int j = 0; j < 2; ++j) {
            const int cidx = wn * 32 + j * 16 + l16;
            bhalf8 c0 = *reinterpret_cast<const bhalf8*>(&Bs[cur][cidx * 40 + halfL * 8]);
            bhalf8 c1 = *reinterpret_cast<const bhalf8*>(&Bs[cur][cidx * 40 + 16 + halfL * 8]);
            bfrag[j] = pack16(c0, c1);
        }

#pragma unroll
        for (int i = 0; i < 2; ++i)
#pragma unroll
            for (int j = 0; j < 2; ++j)
                acc[i][j] = wmma_bf16(afrag[i], bfrag[j], acc[i][j]);
    }

    // ---- epilogue: C layout lane l16 = N col, VGPR g -> M = g + halfL*8 ----
#pragma unroll
    for (int i = 0; i < 2; ++i) {
#pragma unroll
        for (int j = 0; j < 2; ++j) {
            const unsigned int col = bn + wn * 32 + j * 16 + l16;
            const float bval = bias[col];
#pragma unroll
            for (int g = 0; g < 8; ++g) {
                const unsigned int row = bm + wm * 32 + i * 16 + g + halfL * 8;
                const float v = acc[i][j][g] + bval;
                if constexpr (OUTBF16) {
                    ((bhalf*)outv)[row * (unsigned int)N + col] = (bhalf)v;
                } else {
                    ((float*)outv)[row * (unsigned int)N + col] = v;
                }
            }
        }
    }
}

// ---------------------------------------------------------------------------
// Pass A: column softmax statistics.
// One wave per 16-key column tile. The diagonal q-tile (qt == kt) is peeled
// and carries the causal-mask compare chain; all qt > kt tiles are fully
// valid and run a branch-free online (max, sum) update that co-executes with
// the two S-tile WMMAs. Halves combined via shfl_xor 16.
// ---------------------------------------------------------------------------
__global__ __launch_bounds__(256) void attn_colstats_kernel(
    const bhalf* __restrict__ qkv, float* __restrict__ colmax,
    float* __restrict__ colsum) {

    const int lane  = threadIdx.x & 31;
    const int l16   = lane & 15;
    const int halfL = lane >> 4;
    const int wave  = blockIdx.x * 8 + (threadIdx.x >> 5);  // 0..6143

    const int kt = wave & 127;     // key tile 0..127
    const int bh = wave >> 7;      // 0..47
    const int b  = bh / HH;
    const int h  = bh % HH;
    const int kBase = kt * 16;

    const bhalf* qbase = qkv + (size_t)b * TT * C3 + h * HD;
    const bhalf* kbase = qbase + CC;

    // K^T as B operand: lane column n = l16 -> key kBase+l16, per-lane head
    // dims step*32 + halfL*8 + 0..7 and step*32 + 16 + halfL*8 + 0..7. Fixed across q loop.
    bhalf16 kfrag[2];
    {
        const bhalf* krow = kbase + (size_t)(kBase + l16) * C3;
#pragma unroll
        for (int s = 0; s < 2; ++s) {
            bhalf8 c0 = ld8(krow + s * 32 + halfL * 8);
            bhalf8 c1 = ld8(krow + s * 32 + 16 + halfL * 8);
            kfrag[s] = pack16(c0, c1);
        }
    }

    const float scale = 0.125f;  // 1/sqrt(64)
    const int kcol = kBase + l16;

    float mcol, lcol;

    // ---- peeled diagonal tile (qt == kt): the only tile with masking ----
    {
        const bhalf* qrow = qbase + (size_t)(kBase + l16) * C3;
        floatx8 s = zero8();
#pragma unroll
        for (int st = 0; st < 2; ++st) {
            bhalf8 c0 = ld8(qrow + st * 32 + halfL * 8);
            bhalf8 c1 = ld8(qrow + st * 32 + 16 + halfL * 8);
            s = wmma_bf16(pack16(c0, c1), kfrag[st], s);
        }
        float sv[8];
        float tmax = -__builtin_inff();
#pragma unroll
        for (int g = 0; g < 8; ++g) {
            const int q = kBase + g + halfL * 8;
            float x = s[g] * scale;
            if (kcol > q) x = -__builtin_inff();
            sv[g] = x;
            tmax = fmaxf(tmax, x);
        }
        mcol = tmax;  // may be -inf for lanes whose 8 rows are all masked
        lcol = 0.0f;
        if (tmax != -__builtin_inff()) {
#pragma unroll
            for (int g = 0; g < 8; ++g) lcol += __expf(sv[g] - tmax);
        }
    }

    // ---- off-diagonal tiles: no masking, branch-free online update ----
    for (int qt = kt + 1; qt < TT / 16; ++qt) {
        const int qBase = qt * 16;
        const bhalf* qrow = qbase + (size_t)(qBase + l16) * C3;

        floatx8 s = zero8();
#pragma unroll
        for (int st = 0; st < 2; ++st) {
            bhalf8 c0 = ld8(qrow + st * 32 + halfL * 8);
            bhalf8 c1 = ld8(qrow + st * 32 + 16 + halfL * 8);
            s = wmma_bf16(pack16(c0, c1), kfrag[st], s);
        }

        float sv[8];
        float tmax = s[0] * scale;
        sv[0] = tmax;
#pragma unroll
        for (int g = 1; g < 8; ++g) {
            sv[g] = s[g] * scale;
            tmax = fmaxf(tmax, sv[g]);
        }
        const float mnew = fmaxf(mcol, tmax);             // finite
        float l = lcol * __expf(mcol - mnew);             // exp(-inf)=0 is safe
#pragma unroll
        for (int g = 0; g < 8; ++g) l += __expf(sv[g] - mnew);
        lcol = l;
        mcol = mnew;
    }

    // combine the two lane halves (rows 0-7 & 8-15 of each column)
    const float om = __shfl_xor(mcol, 16);
    const float ol = __shfl_xor(lcol, 16);
    const float mm = fmaxf(mcol, om);  // finite: diagonal element always valid
    const float lt = lcol * __expf(mcol - mm) + ol * __expf(om - mm);

    if (lane < 16) {
        colmax[(size_t)bh * TT + kBase + l16] = mm;
        colsum[(size_t)bh * TT + kBase + l16] = lt;
    }
}

// ---------------------------------------------------------------------------
// Pass B: y[q] = sum_{k<=q} exp(s(q,k)-m_k)/l_k * v[k].
// One wave per 16-query tile; block's 8 waves share (b,h) and a
// double-buffered transposed V tile (one barrier per k-step, next V block
// prefetched into registers during compute). Key tiles fully at/below the
// diagonal take a branch-free P path; only straddling tiles run the mask
// compare chain. P goes through per-wave LDS to convert C-layout ->
// A-fragment layout for the P@V WMMAs.
// ---------------------------------------------------------------------------
__global__ __launch_bounds__(256) void attn_out_kernel(
    const bhalf* __restrict__ qkv, const float* __restrict__ colmax,
    const float* __restrict__ colsum, bhalf* __restrict__ y) {

    __shared__ __align__(16) bhalf Vt[2][64 * 40];  // [dim][key], stride 40
    __shared__ __align__(16) bhalf Pt[8][16 * 32];  // per-wave P tile [q][k]

    const int tid   = threadIdx.x;
    const int lane  = tid & 31;
    const int wave  = tid >> 5;
    const int l16   = lane & 15;
    const int halfL = lane >> 4;

    const int bh   = blockIdx.x >> 4;   // 48 (b,h) pairs
    const int qgrp = blockIdx.x & 15;   // 16 blocks per (b,h)
    const int b = bh / HH;
    const int h = bh % HH;

    const int qt    = qgrp * 8 + wave;
    const int qBase = qt * 16;
    const int myKEnd    = qBase + 15;
    const int blockKEnd = (qgrp * 8 + 7) * 16 + 15;

    const bhalf* qbase = qkv + (size_t)b * TT * C3 + h * HD;
    const bhalf* kbase = qbase + CC;
    const bhalf* vbase = qbase + 2 * CC;

    // Q fragment (fixed per wave): rows qBase..qBase+15, head dims 0..63
    bhalf16 qfrag[2];
    {
        const bhalf* qrow = qbase + (size_t)(qBase + l16) * C3;
#pragma unroll
        for (int st = 0; st < 2; ++st) {
            bhalf8 c0 = ld8(qrow + st * 32 + halfL * 8);
            bhalf8 c1 = ld8(qrow + st * 32 + 16 + halfL * 8);
            qfrag[st] = pack16(c0, c1);
        }
    }

    floatx8 yacc[4];
#pragma unroll
    for (int nt = 0; nt < 4; ++nt) yacc[nt] = zero8();

    const int vkk = tid & 31;  // key within tile
    const int vdg = tid >> 5;  // dim group (8 dims each)

    // prologue: preload V block 0 into registers
    bhalf8 vv = ld8(vbase + (size_t)vkk * C3 + vdg * 8);

    int idx = 0;
    for (int kb = 0; kb <= blockKEnd; kb += 32, ++idx) {
        const int cur = idx & 1;

        // ---- stage preloaded V registers, transposed: Vt[cur][dim][key] ----
#pragma unroll
        for (int j = 0; j < 8; ++j) Vt[cur][(vdg * 8 + j) * 40 + vkk] = vv[j];
        __syncthreads();

        // prefetch next V block into registers (overlaps compute)
        if (kb + 32 <= blockKEnd) {
            vv = ld8(vbase + (size_t)(kb + 32 + vkk) * C3 + vdg * 8);
        }

        if (kb <= myKEnd) {  // wave-uniform: EXEC stays full for WMMA
            // ---- S and P for two 16-key column tiles ----
#pragma unroll
            for (int kt2 = 0; kt2 < 2; ++kt2) {
                const int kcb = kb + kt2 * 16;
                const bhalf* krow = kbase + (size_t)(kcb + l16) * C3;

                floatx8 s = zero8();
#pragma unroll
                for (int st = 0; st < 2; ++st) {
                    bhalf8 c0 = ld8(krow + st * 32 + halfL * 8);
                    bhalf8 c1 = ld8(krow + st * 32 + 16 + halfL * 8);
                    s = wmma_bf16(qfrag[st], pack16(c0, c1), s);
                }

                const int kcol = kcb + l16;
                const float cm  = colmax[(size_t)bh * TT + kcol];
                const float inv = 1.0f / colsum[(size_t)bh * TT + kcol];

                if (kcb + 15 <= qBase) {
                    // tile fully at/below diagonal: branch-free P
#pragma unroll
                    for (int g = 0; g < 8; ++g) {
                        const float p = __expf(s[g] * 0.125f - cm) * inv;
                        Pt[wave][(g + halfL * 8) * 32 + kt2 * 16 + l16] = (bhalf)p;
                    }
                } else {
                    // tile straddles the diagonal: per-element causal mask
#pragma unroll
                    for (int g = 0; g < 8; ++g) {
                        const int q = qBase + g + halfL * 8;
                        const float p =
                            (kcol <= q) ? __expf(s[g] * 0.125f - cm) * inv : 0.0f;
                        Pt[wave][(g + halfL * 8) * 32 + kt2 * 16 + l16] = (bhalf)p;
                    }
                }
            }

            // ---- P as A-fragment (16 q rows x 32 keys) ----
            bhalf8 p0 = *reinterpret_cast<const bhalf8*>(&Pt[wave][l16 * 32 + halfL * 8]);
            bhalf8 p1 = *reinterpret_cast<const bhalf8*>(&Pt[wave][l16 * 32 + 16 + halfL * 8]);
            const bhalf16 pfrag = pack16(p0, p1);

            // ---- P @ V : 4 n-tiles of 16 head dims ----
#pragma unroll
            for (int nt = 0; nt < 4; ++nt) {
                const int d = nt * 16 + l16;
                bhalf8 c0 = *reinterpret_cast<const bhalf8*>(&Vt[cur][d * 40 + halfL * 8]);
                bhalf8 c1 = *reinterpret_cast<const bhalf8*>(&Vt[cur][d * 40 + 16 + halfL * 8]);
                yacc[nt] = wmma_bf16(pfrag, pack16(c0, c1), yacc[nt]);
            }
        }
    }

    // ---- store y tile (bf16, merged-head layout [B,T,C]) ----
#pragma unroll
    for (int nt = 0; nt < 4; ++nt) {
#pragma unroll
        for (int g = 0; g < 8; ++g) {
            const int q = qBase + g + halfL * 8;
            const int d = nt * 16 + l16;
            y[((size_t)(b * TT + q)) * CC + h * HD + d] = (bhalf)yacc[nt][g];
        }
    }
}

// ---------------------------------------------------------------------------
// Launch
// ---------------------------------------------------------------------------
extern "C" void kernel_launch(void* const* d_in, const int* in_sizes, int n_in,
                              void* d_out, int out_size, void* d_ws, size_t ws_size,
                              hipStream_t stream) {
    const float* x      = (const float*)d_in[0];  // [B,T,C]
    const float* w_attn = (const float*)d_in[1];  // [C,3C]
    const float* b_attn = (const float*)d_in[2];  // [3C]
    const float* w_proj = (const float*)d_in[3];  // [C,C]
    const float* b_proj = (const float*)d_in[4];  // [C]
    float* out = (float*)d_out;                   // [B,T,C]

    // workspace layout (bf16 buffers + f32 stats), 256B-aligned regions
    char* ws = (char*)d_ws;
    size_t off = 0;
    auto take = [&](size_t bytes) {
        char* p = ws + off;
        off = (off + bytes + 255) & ~(size_t)255;
        return p;
    };
    bhalf* xb    = (bhalf*)take((size_t)MM * CC * 2);
    bhalf* wab   = (bhalf*)take((size_t)CC * C3 * 2);
    bhalf* wpb   = (bhalf*)take((size_t)CC * CC * 2);
    bhalf* qkvb  = (bhalf*)take((size_t)MM * C3 * 2);
    bhalf* yb    = (bhalf*)take((size_t)MM * CC * 2);
    float* cmax  = (float*)take((size_t)BB * HH * TT * 4);
    float* csum  = (float*)take((size_t)BB * HH * TT * 4);
    (void)ws_size; (void)in_sizes; (void)n_in; (void)out_size;

    // 1. fp32 -> bf16 conversions
    cvt_f32_bf16_kernel<<<1024, 256, 0, stream>>>(x, xb, MM * CC);
    cvt_f32_bf16_kernel<<<512, 256, 0, stream>>>(w_attn, wab, CC * C3);
    cvt_f32_bf16_kernel<<<256, 256, 0, stream>>>(w_proj, wpb, CC * CC);

    // 2. qkv = x @ w_attn + b_attn  -> bf16 [M, 3C]
    {
        dim3 grid(C3 / 64, MM / 128);
        gemm_bf16_kernel<C3, CC, true><<<grid, 256, 0, stream>>>(xb, wab, b_attn, qkvb);
    }

    // 3. column-softmax statistics (max/sum over queries per key column)
    attn_colstats_kernel<<<(BB * HH * (TT / 16)) / 8, 256, 0, stream>>>(
        qkvb, cmax, csum);

    // 4. attention output y (bf16, merged heads)
    attn_out_kernel<<<BB * HH * 16, 256, 0, stream>>>(qkvb, cmax, csum, yb);

    // 5. out = y @ w_proj + b_proj  -> fp32
    {
        dim3 grid(CC / 64, MM / 128);
        gemm_bf16_kernel<CC, CC, false><<<grid, 256, 0, stream>>>(yb, wpb, b_proj, out);
    }
}
